// Extractor_51393578664113
// MI455X (gfx1250) — matrix operand order repaired and checked by reading, not compiled, not run on hardware
//
#include <hip/hip_runtime.h>
#include <hip/hip_bf16.h>

// ---------------- fixed problem constants (from reference setup) ----------------
#define BB      2
#define HH      48
#define WW      48
#define DD      768
#define LL      (HH*WW)            // 2304
#define NN      12096              // 21 * (HH*WW/4)
#define M1      (BB*NN)            // 24192 query rows
#define MLr     (BB*LL)            // 4608 feat rows
#define HEADS   6
#define DH      128
#define PP      4
#define HID     192
#define NSMALL  576                // NN/21
#define NOFF    80                 // 48 off + 24 attn, padded to 80 (5 x 16)

typedef _Float16 h4  __attribute__((ext_vector_type(4)));
typedef _Float16 h8  __attribute__((ext_vector_type(8)));
typedef _Float16 v16h __attribute__((ext_vector_type(16)));
typedef float    v8f  __attribute__((ext_vector_type(8)));

// ---------------- LayerNorm (f32 in, f16 out), one block per row of 768 ----------------
__global__ void ln_f16_kernel(const float* __restrict__ x, const float* __restrict__ g,
                              const float* __restrict__ bta, _Float16* __restrict__ out)
{
    const int row = blockIdx.x;
    const float* xr = x + (size_t)row * DD;
    float s = 0.f, s2 = 0.f;
    for (int i = threadIdx.x; i < DD; i += 256) { float v = xr[i]; s += v; s2 += v * v; }
    for (int off = 16; off > 0; off >>= 1) {
        s  += __shfl_down(s,  off, 32);
        s2 += __shfl_down(s2, off, 32);
    }
    __shared__ float sh[16];
    const int wv = threadIdx.x >> 5;
    if ((threadIdx.x & 31) == 0) { sh[wv] = s; sh[8 + wv] = s2; }
    __syncthreads();
    if (threadIdx.x == 0) {
        float ts = 0.f, ts2 = 0.f;
        for (int w = 0; w < 8; ++w) { ts += sh[w]; ts2 += sh[8 + w]; }
        float mean = ts * (1.f / DD);
        float var  = ts2 * (1.f / DD) - mean * mean;
        sh[0] = mean; sh[1] = rsqrtf(var + 1e-6f);
    }
    __syncthreads();
    const float mean = sh[0], rstd = sh[1];
    _Float16* orow = out + (size_t)row * DD;
    for (int i = threadIdx.x; i < DD; i += 256)
        orow[i] = (_Float16)(((xr[i] - mean) * rstd) * g[i] + bta[i]);
}

// ---------------- weight transpose+convert: W[K][Nc] f32 -> Wt[Nrows][K] f16 ----------------
__global__ void wT_f16_kernel(const float* __restrict__ Wsrc, _Float16* __restrict__ Wt,
                              int K, int Ncols, int Nrows)
{
    const size_t idx = (size_t)blockIdx.x * 256 + threadIdx.x;
    const size_t tot = (size_t)Nrows * K;
    if (idx >= tot) return;
    const int c = (int)(idx / K);
    const int k = (int)(idx % K);
    float v = (c < Ncols) ? Wsrc[(size_t)k * Ncols + c] : 0.f;
    Wt[idx] = (_Float16)v;
}

// ---------------- pack Woff(768x48)+Wattn(768x24) -> Wt[80][768] f16 ----------------
__global__ void pack_offattn_w(const float* __restrict__ Woff, const float* __restrict__ Wattn,
                               _Float16* __restrict__ Wt)
{
    const int idx = blockIdx.x * 256 + threadIdx.x;
    if (idx >= NOFF * DD) return;
    const int c = idx / DD, k = idx % DD;
    float v = 0.f;
    if (c < 48)      v = Woff [(size_t)k * 48 + c];
    else if (c < 72) v = Wattn[(size_t)k * 24 + (c - 48)];
    Wt[idx] = (_Float16)v;
}
__global__ void pack_offattn_b(const float* __restrict__ boff, const float* __restrict__ battn,
                               float* __restrict__ bias)
{
    const int c = threadIdx.x;
    if (c >= NOFF) return;
    float v = 0.f;
    if (c < 48) v = boff[c]; else if (c < 72) v = battn[c - 48];
    bias[c] = v;
}

// ---------------- WMMA GEMM: out[M][ldo] = A[M][KVAL] @ Bt[*][KVAL]^T (+bias)(+res) ----------------
// 8 waves / block; each wave owns one 16(M) x (NT*16)(N) tile, K stepped by 32 via
// v_wmma_f32_16x16x32_f16. All control decisions are compile-time so the K-loop is branch-free
// and EXEC stays all-ones (WMMA requirement).
// A fragment (ISA 7.12.2): lane%16 = M row; lane/16 selects K 0-7/8-15 (VGPR0-3), 16-23/24-31 (VGPR4-7).
// B fragment: lane%16 = N col; same K order as A: lane/16 selects K 0-7/8-15 (elems 0-7), 16-23/24-31 (elems 8-15)
//             -> two 16-byte loads from the transposed f16 weights.
template <int KVAL, int NT, bool HAS_BIAS, bool HAS_RES, typename OutT>
__global__ void gemm_wmma(const _Float16* __restrict__ A, const _Float16* __restrict__ Bt,
                          const float* __restrict__ bias, const float* __restrict__ res,
                          OutT* __restrict__ out, int ldo, int Nstore)
{
    const int lane = threadIdx.x & 31;
    const int wv   = threadIdx.x >> 5;
    const int lm   = lane & 15;
    const int hi   = lane >> 4;                              // 0 or 1
    const int row0 = (blockIdx.x * 8 + wv) * 16;
    const int nt0  = blockIdx.y * NT;

    v8f acc[NT] = {};
    const _Float16* ap = A + (size_t)(row0 + lm) * KVAL + hi * 8;
    const _Float16* bp[NT];
#pragma unroll
    for (int t = 0; t < NT; ++t)
        bp[t] = Bt + (size_t)((nt0 + t) * 16 + lm) * KVAL + hi * 8;

#pragma unroll 4
    for (int kk = 0; kk < KVAL; kk += 32) {
        h8 a_lo = *(const h8*)(ap);
        h8 a_hi = *(const h8*)(ap + 16);
        v16h af = __builtin_shufflevector(a_lo, a_hi,
                                          0,1,2,3,4,5,6,7,8,9,10,11,12,13,14,15);
#pragma unroll
        for (int t = 0; t < NT; ++t) {
            v16h bf = __builtin_shufflevector(*(const h8*)(bp[t]), *(const h8*)(bp[t] + 16),
                                              0,1,2,3,4,5,6,7,8,9,10,11,12,13,14,15);
            acc[t] = __builtin_amdgcn_wmma_f32_16x16x32_f16(
                        false, af, false, bf, (short)0, acc[t], false, false);
            bp[t] += 32;
        }
        ap += 32;
    }

#pragma unroll
    for (int t = 0; t < NT; ++t) {
        const int col = (nt0 + t) * 16 + lm;
        if (col < Nstore) {
            float bv = 0.f;
            if (HAS_BIAS) bv = bias[col];
#pragma unroll
            for (int r = 0; r < 8; ++r) {
                const int row = row0 + r + hi * 8;           // C/D layout: VGPR r -> M r / r+8
                float v = acc[t][r] + bv;
                if (HAS_RES) v += res[(size_t)row * ldo + col];
                out[(size_t)row * ldo + col] = (OutT)v;
            }
        }
    }
}

// ---------------- deformable sampling: one wave per (b, q, head) ----------------
__global__ void sample_kernel(const float* __restrict__ refp, const float* __restrict__ offaw,
                              const _Float16* __restrict__ value, _Float16* __restrict__ attn_in)
{
    const int gtid = blockIdx.x * blockDim.x + threadIdx.x;
    const int wid  = gtid >> 5;
    const int lane = gtid & 31;
    if (wid >= BB * NN * HEADS) return;
    const int head = wid % HEADS;
    const int bq   = wid / HEADS;               // b*NN + q
    const int b    = bq / NN;

    const float rx = refp[(size_t)bq * 2 + 0];
    const float ry = refp[(size_t)bq * 2 + 1];
    const float* oa = offaw + (size_t)bq * NOFF;

    // softmax over P=4 attention logits
    float lg[PP]; float mx = -1e30f;
#pragma unroll
    for (int p = 0; p < PP; ++p) { lg[p] = oa[48 + head * PP + p]; mx = fmaxf(mx, lg[p]); }
    float se = 0.f;
#pragma unroll
    for (int p = 0; p < PP; ++p) { lg[p] = __expf(lg[p] - mx); se += lg[p]; }
    const float inv = 1.f / se;

    float acc0 = 0.f, acc1 = 0.f, acc2 = 0.f, acc3 = 0.f;
    const int d0 = lane * 4;                    // dh = 128 = 32 lanes * 4
    const _Float16* vb = value + ((size_t)b * LL) * DD + head * DH + d0;

#pragma unroll
    for (int p = 0; p < PP; ++p) {
        const float ox = oa[head * (PP * 2) + p * 2 + 0];
        const float oy = oa[head * (PP * 2) + p * 2 + 1];
        const float xf = (rx + ox * (1.f / WW)) * WW - 0.5f;
        const float yf = (ry + oy * (1.f / HH)) * HH - 0.5f;
        const float x0f = floorf(xf), y0f = floorf(yf);
        const int x0 = (int)x0f, y0 = (int)y0f;
        const float wx = xf - x0f, wy = yf - y0f;
        const float wgt = lg[p] * inv;
        const float cw[4] = { (1.f-wx)*(1.f-wy), wx*(1.f-wy), (1.f-wx)*wy, wx*wy };
        const int cx[4] = { x0, x0+1, x0,   x0+1 };
        const int cy[4] = { y0, y0,   y0+1, y0+1 };
#pragma unroll
        for (int c = 0; c < 4; ++c) {
            const bool valid = (cx[c] >= 0) & (cx[c] < WW) & (cy[c] >= 0) & (cy[c] < HH);
            const int xi = min(max(cx[c], 0), WW - 1);
            const int yi = min(max(cy[c], 0), HH - 1);
            const float w = valid ? wgt * cw[c] : 0.f;
            const h4 vv = *(const h4*)(vb + (size_t)(yi * WW + xi) * DD);
            acc0 += w * (float)vv[0]; acc1 += w * (float)vv[1];
            acc2 += w * (float)vv[2]; acc3 += w * (float)vv[3];
        }
    }
    h4 o; o[0] = (_Float16)acc0; o[1] = (_Float16)acc1; o[2] = (_Float16)acc2; o[3] = (_Float16)acc3;
    *(h4*)(attn_in + (size_t)bq * DD + head * DH + d0) = o;
}

// ---------------- depthwise 3x3 conv (SAME, zero pad) + exact GELU ----------------
__global__ void dwconv_gelu_kernel(const float* __restrict__ hb, const float* __restrict__ dww,
                                   const float* __restrict__ dwb, _Float16* __restrict__ out)
{
    const size_t idx = (size_t)blockIdx.x * 256 + threadIdx.x;
    if (idx >= (size_t)BB * NN * HID) return;
    const int c = (int)(idx % HID);
    const int t = (int)((idx / HID) % NN);
    const int b = (int)(idx / ((size_t)HID * NN));

    int base, hh, ww, tt;
    if (t < 16 * NSMALL)      { base = 0;           hh = 2*HH; ww = 2*WW; tt = t; }
    else if (t < 20 * NSMALL) { base = 16 * NSMALL; hh = HH;   ww = WW;   tt = t - 16*NSMALL; }
    else                      { base = 20 * NSMALL; hh = HH/2; ww = WW/2; tt = t - 20*NSMALL; }
    const int y = tt / ww, x = tt % ww;

    const float* w9 = dww + (size_t)c * 9;
    float s = dwb[c];
#pragma unroll
    for (int dy = -1; dy <= 1; ++dy) {
#pragma unroll
        for (int dx = -1; dx <= 1; ++dx) {
            const int yy = y + dy, xx = x + dx;
            if (yy >= 0 && yy < hh && xx >= 0 && xx < ww)
                s += w9[(dy+1)*3 + (dx+1)] *
                     hb[((size_t)b * NN + base + yy * ww + xx) * HID + c];
        }
    }
    const float g = 0.5f * s * (1.f + erff(s * 0.70710678118654752f)); // exact GELU
    out[idx] = (_Float16)g;
}

// ---------------- host-side orchestration ----------------
extern "C" void kernel_launch(void* const* d_in, const int* in_sizes, int n_in,
                              void* d_out, int out_size, void* d_ws, size_t ws_size,
                              hipStream_t stream)
{
    const float* query = (const float*)d_in[0];
    const float* refp  = (const float*)d_in[1];
    const float* feat  = (const float*)d_in[2];
    const float* qn_g  = (const float*)d_in[7];
    const float* qn_b  = (const float*)d_in[8];
    const float* fn_g  = (const float*)d_in[9];
    const float* fn_b  = (const float*)d_in[10];
    const float* Wv    = (const float*)d_in[11];
    const float* bv    = (const float*)d_in[12];
    const float* Woff  = (const float*)d_in[13];
    const float* boff  = (const float*)d_in[14];
    const float* Wattn = (const float*)d_in[15];
    const float* battn = (const float*)d_in[16];
    const float* Wo    = (const float*)d_in[17];
    const float* bo    = (const float*)d_in[18];
    const float* ffn_g = (const float*)d_in[19];
    const float* ffn_b = (const float*)d_in[20];
    const float* W1    = (const float*)d_in[21];
    const float* b1    = (const float*)d_in[22];
    const float* dw_w  = (const float*)d_in[23];
    const float* dw_b  = (const float*)d_in[24];
    const float* W2    = (const float*)d_in[25];
    const float* b2    = (const float*)d_in[26];
    float* out = (float*)d_out;

    // workspace carve-up (~156 MB total), 256B aligned, sequential reuse of bufA
    char* ws = (char*)d_ws;
    size_t o = 0;
    auto alloc = [&](size_t bytes) { size_t r = o; o = (o + bytes + 255) & ~(size_t)255; return r; };
    const size_t oBufA  = alloc((size_t)M1 * DD * 2);   // qn_f16 -> attn_in_f16 -> ln2_f16 -> h2_f16
    const size_t oFn16  = alloc((size_t)MLr * DD * 2);
    const size_t oVal16 = alloc((size_t)MLr * DD * 2);
    const size_t oWvT   = alloc((size_t)DD * DD * 2);
    const size_t oWoT   = alloc((size_t)DD * DD * 2);
    const size_t oW1T   = alloc((size_t)HID * DD * 2);
    const size_t oW2T   = alloc((size_t)DD * HID * 2);
    const size_t oWofT  = alloc((size_t)NOFF * DD * 2);
    const size_t oBias  = alloc((size_t)NOFF * 4);
    const size_t oOffaw = alloc((size_t)M1 * NOFF * 4);
    const size_t oQ2    = alloc((size_t)M1 * DD * 4);
    const size_t oHbuf  = alloc((size_t)M1 * HID * 4);
    (void)ws_size;

    _Float16* bufA   = (_Float16*)(ws + oBufA);
    _Float16* fn16   = (_Float16*)(ws + oFn16);
    _Float16* val16  = (_Float16*)(ws + oVal16);
    _Float16* WvT    = (_Float16*)(ws + oWvT);
    _Float16* WoT    = (_Float16*)(ws + oWoT);
    _Float16* W1T    = (_Float16*)(ws + oW1T);
    _Float16* W2T    = (_Float16*)(ws + oW2T);
    _Float16* WofT   = (_Float16*)(ws + oWofT);
    float*    bias80 = (float*)(ws + oBias);
    float*    offaw  = (float*)(ws + oOffaw);
    float*    q2     = (float*)(ws + oQ2);
    float*    hbuf   = (float*)(ws + oHbuf);

    // 1) LayerNorms -> f16
    ln_f16_kernel<<<M1,  256, 0, stream>>>(query, qn_g, qn_b, bufA);
    ln_f16_kernel<<<MLr, 256, 0, stream>>>(feat,  fn_g, fn_b, fn16);

    // 2) weight conversion (f32 -> f16 transposed)
    wT_f16_kernel<<<(DD*DD + 255)/256, 256, 0, stream>>>(Wv, WvT, DD, DD, DD);
    wT_f16_kernel<<<(DD*DD + 255)/256, 256, 0, stream>>>(Wo, WoT, DD, DD, DD);
    wT_f16_kernel<<<(HID*DD + 255)/256, 256, 0, stream>>>(W1, W1T, DD, HID, HID);
    wT_f16_kernel<<<(DD*HID + 255)/256, 256, 0, stream>>>(W2, W2T, HID, DD, DD);
    pack_offattn_w<<<(NOFF*DD + 255)/256, 256, 0, stream>>>(Woff, Wattn, WofT);
    pack_offattn_b<<<1, 128, 0, stream>>>(boff, battn, bias80);

    // 3) value = LN(feat) @ Wv + bv   (f16 out for sampling)
    gemm_wmma<DD, 4, true, false, _Float16><<<dim3(MLr/128, DD/64), 256, 0, stream>>>(
        fn16, WvT, bv, nullptr, val16, DD, DD);

    // 4) off / attn logits = qn @ [Woff|Wattn] + bias  (80 padded cols, 5 n-tiles in one wave)
    gemm_wmma<DD, 5, true, false, float><<<dim3(M1/128, 1), 256, 0, stream>>>(
        bufA, WofT, bias80, nullptr, offaw, NOFF, NOFF);

    // 5) deformable bilinear sampling -> attn_in (f16, reuses bufA)
    sample_kernel<<<(BB*NN*HEADS)/8, 256, 0, stream>>>(refp, offaw, val16, bufA);

    // 6) query2 = query + attn_in @ Wo + bo
    gemm_wmma<DD, 4, true, true, float><<<dim3(M1/128, DD/64), 256, 0, stream>>>(
        bufA, WoT, bo, query, q2, DD, DD);

    // 7) LN(query2) -> f16 (reuses bufA)
    ln_f16_kernel<<<M1, 256, 0, stream>>>(q2, ffn_g, ffn_b, bufA);

    // 8) h = ln2 @ W1 + b1
    gemm_wmma<DD, 4, true, false, float><<<dim3(M1/128, HID/64), 256, 0, stream>>>(
        bufA, W1T, b1, nullptr, hbuf, HID, HID);

    // 9) depthwise 3x3 conv + GELU -> h2 (f16, reuses bufA)
    dwconv_gelu_kernel<<<(int)(((size_t)BB*NN*HID + 255)/256), 256, 0, stream>>>(
        hbuf, dw_w, dw_b, bufA);

    // 10) out = query2 + h2 @ W2 + b2
    gemm_wmma<HID, 4, true, true, float><<<dim3(M1/128, DD/64), 256, 0, stream>>>(
        bufA, W2T, b2, q2, out, DD, DD);
}
